// LSTM_EMG_43628277793434
// MI455X (gfx1250) — matrix operand order repaired and checked by reading, not compiled, not run on hardware
//
#include <hip/hip_runtime.h>

#define DI static __device__ __forceinline__

typedef __attribute__((ext_vector_type(16))) __bf16   v16bf;
typedef __attribute__((ext_vector_type(8)))  float    v8f;
typedef __attribute__((ext_vector_type(4)))  float    v4f;
typedef __attribute__((ext_vector_type(4)))  unsigned v4u;

struct U8x32 { v4u lo, hi; };   // 32 bytes -> one v16bf fragment
struct F4x2  { v4f lo, hi; };   // 32 bytes -> one v8f accumulator

DI v16bf mkfrag(v4u lo, v4u hi) { U8x32 t; t.lo = lo; t.hi = hi; return __builtin_bit_cast(v16bf, t); }

DI unsigned short f2bf(float f) {              // f32 -> bf16, round-to-nearest-even
  unsigned u = __builtin_bit_cast(unsigned, f);
  return (unsigned short)((u + 0x7fffu + ((u >> 16) & 1u)) >> 16);
}
DI float bf2f(unsigned short v) { return __builtin_bit_cast(float, ((unsigned)v) << 16); }

DI float fast_rcp(float x) {
#if __has_builtin(__builtin_amdgcn_rcpf)
  return __builtin_amdgcn_rcpf(x);           // raw v_rcp_f32, no div fixup
#else
  return 1.0f / x;
#endif
}
DI float tanh_f(float x) {
#if __has_builtin(__builtin_amdgcn_tanhf)
  return __builtin_amdgcn_tanhf(x);          // single v_tanh_f32 trans op
#else
  return 1.0f - 2.0f * fast_rcp(1.0f + __expf(2.0f * x));
#endif
}
DI float sigm(float x) { return 0.5f * tanh_f(0.5f * x) + 0.5f; }   // 1 trans + fma

constexpr int T  = 2048, NB = 512, IN = 16, H1 = 5, H2 = 50, NC = 20;
constexpr int G1 = 4 * H1;     // 20 gates, layer 1
constexpr int XPR = 24;        // padded xp1 row length (bf16 elems, 48B)
constexpr int KD  = 64;        // padded K for layer-2 WMMA (2 chunks of 32)
constexpr int NGP = 256;       // padded gate dim (16 N-tiles of 16 -> 4/wave)
constexpr int ABUF = 4 * 16 * KD;   // A staging buffer, elems (one of two)

// ---------------------------------------------------------------------------
// K0: xp1[t][b][g] = x[b][t][:] @ W_ih1.T + (b_ih1 + b_hh1)   (bf16 output)
// ---------------------------------------------------------------------------
__global__ void __launch_bounds__(256, 1)
k0_inproj(const float* __restrict__ x,
          const float* __restrict__ Wih1,
          const float* __restrict__ bih1,
          const float* __restrict__ bhh1,
          unsigned short* __restrict__ xp1) {
  __shared__ float w[G1 * IN];
  __shared__ float bs[G1];
  const int tid = threadIdx.x;
  for (int i = tid; i < G1 * IN; i += 256) w[i] = Wih1[i];
  if (tid < G1) bs[tid] = bih1[tid] + bhh1[tid];
  __syncthreads();

  size_t r = (size_t)blockIdx.x * 256 + tid;   // r = b*T + t
  const int b = (int)(r >> 11);
  const int t = (int)(r & 2047);

  const v4f* xr = (const v4f*)(x + r * IN);
  v4f x0 = xr[0], x1 = xr[1], x2 = xr[2], x3 = xr[3];
  float xv[16];
#pragma unroll
  for (int i = 0; i < 4; ++i) { xv[i] = x0[i]; xv[4+i] = x1[i]; xv[8+i] = x2[i]; xv[12+i] = x3[i]; }

  unsigned out[12];
#pragma unroll
  for (int g = 0; g < G1; g += 2) {
    float s0 = bs[g], s1 = bs[g + 1];
#pragma unroll
    for (int i = 0; i < IN; ++i) {
      s0 += xv[i] * w[g * IN + i];
      s1 += xv[i] * w[(g + 1) * IN + i];
    }
    out[g >> 1] = (unsigned)f2bf(s0) | ((unsigned)f2bf(s1) << 16);
  }
  out[10] = 0u; out[11] = 0u;

  v4u* dst = (v4u*)(xp1 + ((size_t)t * NB + b) * XPR);
  v4u o0 = {out[0], out[1], out[2], out[3]};
  v4u o1 = {out[4], out[5], out[6], out[7]};
  v4u o2 = {out[8], out[9], out[10], out[11]};
  dst[0] = o0; dst[1] = o1; dst[2] = o2;
}

// ---------------------------------------------------------------------------
// K1: layer-1 LSTM scan. One thread per batch element; W_hh1 in VGPRs.
// Emits hs1e[t][b][8] = {h0..h4 (bf16), 1.0bf, 0, 0}.
// ---------------------------------------------------------------------------
__global__ void __launch_bounds__(128, 1)
k1_scan1(const float* __restrict__ Whh1,
         const unsigned short* __restrict__ xp1,
         unsigned short* __restrict__ hs1e) {
  const int b = blockIdx.x * 128 + threadIdx.x;
  float whh[G1 * H1];
#pragma unroll
  for (int i = 0; i < G1 * H1; ++i) whh[i] = Whh1[i];

  float h[H1], c[H1];
#pragma unroll
  for (int j = 0; j < H1; ++j) { h[j] = 0.0f; c[j] = 0.0f; }

  for (int t = 0; t < T; ++t) {
    const v4u* xr = (const v4u*)(xp1 + ((size_t)t * NB + b) * XPR);
    v4u w0 = xr[0], w1 = xr[1], w2 = xr[2];
    float xp[G1];
#pragma unroll
    for (int k = 0; k < 4; ++k) {
      xp[2*k]     = bf2f((unsigned short)(w0[k] & 0xffffu));
      xp[2*k + 1] = bf2f((unsigned short)(w0[k] >> 16));
      xp[8 + 2*k]     = bf2f((unsigned short)(w1[k] & 0xffffu));
      xp[8 + 2*k + 1] = bf2f((unsigned short)(w1[k] >> 16));
    }
#pragma unroll
    for (int k = 0; k < 2; ++k) {
      xp[16 + 2*k]     = bf2f((unsigned short)(w2[k] & 0xffffu));
      xp[16 + 2*k + 1] = bf2f((unsigned short)(w2[k] >> 16));
    }

    float g[G1];
#pragma unroll
    for (int gg = 0; gg < G1; ++gg) {
      float s = xp[gg];
#pragma unroll
      for (int j = 0; j < H1; ++j) s += h[j] * whh[gg * H1 + j];
      g[gg] = s;
    }
#pragma unroll
    for (int j = 0; j < H1; ++j) {     // gate order: i, f, g, o
      float cn = sigm(g[H1 + j]) * c[j] + sigm(g[j]) * tanh_f(g[2 * H1 + j]);
      c[j] = cn;
      h[j] = sigm(g[3 * H1 + j]) * tanh_f(cn);
    }
    unsigned r0 = (unsigned)f2bf(h[0]) | ((unsigned)f2bf(h[1]) << 16);
    unsigned r1 = (unsigned)f2bf(h[2]) | ((unsigned)f2bf(h[3]) << 16);
    unsigned r2 = (unsigned)f2bf(h[4]) | (0x3F80u << 16);      // 1.0 bias column
    v4u o = {r0, r1, r2, 0u};
    *(v4u*)(hs1e + ((size_t)t * NB + b) * 8) = o;
  }
}

// ---------------------------------------------------------------------------
// K2: layer-2 LSTM scan with WMMA.  8 WGs x 512 thr; WG owns 64 batch rows.
// Per step: gates[64x256] = A[64x64] x B[64x256] via v_wmma_f32_16x16x32_bf16.
// - B columns gate-interleaved (n' = 4*j + gate), activation fused in-register.
// - N-tiles interleaved across waves (tile = q + 4*i) so each wave's cell
//   update consumes only its OWN gate tiles -> gate LDS is wave-private,
//   no barrier between GEMM and cell update (same-wave LDS ordering).
// - A staging double-buffered: step t reads buf[t&1], writes buf[(t+1)&1]
//   -> single barrier per step.
// - hs1 row for t+1 prefetched from global at the top of step t.
// ---------------------------------------------------------------------------
__global__ void __launch_bounds__(512, 1)
k2_scan2(const float* __restrict__ Wih2,
         const float* __restrict__ Whh2,
         const float* __restrict__ bih2,
         const float* __restrict__ bhh2,
         const unsigned short* __restrict__ hs1e,
         float* __restrict__ hfin) {
  extern __shared__ char smem[];
  unsigned short* Bw = (unsigned short*)smem;                       // [NGP][KD] bf16 (32768 B)
  unsigned short* Ab = (unsigned short*)(smem + NGP * KD * 2);      // 2x[4][16][KD] bf16 (16384 B)
  float*         gsh = (float*)(smem + NGP * KD * 2 + 2 * ABUF * 2);// [4][NGP][16] f32 (65536 B)

  const int tid   = threadIdx.x;
  const int lane  = tid & 31;
  const int wv    = tid >> 5;
  const int mt    = wv & 3;     // M-tile (16 batch rows)
  const int q     = wv >> 2;    // N slot: owns tiles q, q+4, q+8, q+12
  const int bbase = blockIdx.x * 64;
  const bool stager = (q == 0) && (lane < 16);

  // zero both A staging buffers (covers all K padding; h_prev starts at 0)
  for (int i = tid; i < (2 * ABUF) / 2; i += 512) ((unsigned*)Ab)[i] = 0u;

  // build B, columns gate-interleaved: n' = 4*j + gt  ->  source row gt*50+j
  if (tid < NGP) {
    const int np = tid;
    unsigned short* row = Bw + np * KD;
    if (np < 4 * H2) {
      const int j  = np >> 2;          // hidden index 0..49
      const int gt = np & 3;           // 0=i 1=f 2=g 3=o
      const int gs = gt * H2 + j;      // source gate row in [0,200)
#pragma unroll
      for (int k = 0; k < H1; ++k) row[k] = f2bf(Wih2[gs * H1 + k]);
      row[5] = f2bf(bih2[gs] + bhh2[gs]);
      row[6] = 0; row[7] = 0;
      for (int jj = 0; jj < H2; ++jj) row[8 + jj] = f2bf(Whh2[gs * H2 + jj]);
#pragma unroll
      for (int k = 8 + H2; k < KD; ++k) row[k] = 0;
    } else {
      for (int k = 0; k < KD; ++k) row[k] = 0;
    }
  }
  __syncthreads();                     // zero-init + B build visible

  // stage hs1 row for t=0 into buffer 0
  if (stager) {
    const v4u s0 = *(const v4u*)(hs1e + ((size_t)0 * NB + bbase + mt * 16 + lane) * 8);
    *(v4u*)(Ab + (mt * 16 + lane) * KD) = s0;
  }

  const int col  = lane & 15;
  const int hi16 = (lane & 16) ? 1 : 0;
  const int lowK = hi16 ? 8 : 0;              // A fragment K-pair offset
  const int klo  = hi16 ? 8 : 0;              // B fragment K offset
  const int r0   = hi16 ? 8 : 0;              // accumulator row base

  // per-lane activation constants: sigmoid = 0.5*tanh(0.5x)+0.5; cand = tanh(x)
  const int gtype = col & 3;
  const float s_act = (gtype == 2) ? 1.0f : 0.5f;
  const float a_act = (gtype == 2) ? 1.0f : 0.5f;
  const float b_act = (gtype == 2) ? 0.0f : 0.5f;

  // Preload all 8 B fragments (4 interleaved N-tiles x 2 K-chunks) into VGPRs.
  v16bf bfr[8];
#pragma unroll
  for (int i = 0; i < 4; ++i) {
    const int n = (q + 4 * i) * 16 + col;
    const unsigned short* bp = Bw + n * KD;
    bfr[2*i]     = mkfrag(*(const v4u*)(bp + klo),      *(const v4u*)(bp + klo + 16));
    bfr[2*i + 1] = mkfrag(*(const v4u*)(bp + 32 + klo), *(const v4u*)(bp + 32 + klo + 16));
  }
  __syncthreads();                     // t=0 staging visible to all waves

  float c[8];
#pragma unroll
  for (int i = 0; i < 8; ++i) c[i] = 0.0f;

  const int apOff = (mt * 16 + col) * KD;
  const float* gm = gsh + mt * NGP * 16;

  for (int t = 0; t < T; ++t) {
    // prefetch hs1 row for t+1 (consumed at the bottom of this step)
    v4u nxt = {0u, 0u, 0u, 0u};
    if (stager && (t + 1 < T)) {
      nxt = *(const v4u*)(hs1e + ((size_t)(t + 1) * NB + bbase + mt * 16 + lane) * 8);
    }

    const unsigned short* bufA = Ab + (t & 1) * ABUF;          // read buffer
    unsigned short*       bufB = Ab + ((t + 1) & 1) * ABUF;    // write buffer

    // (b) fused gate GEMM + in-register activation
    const unsigned short* ap = bufA + apOff;
    v16bf a0 = mkfrag(*(const v4u*)(ap + lowK),      *(const v4u*)(ap + lowK + 16));
    v16bf a1 = mkfrag(*(const v4u*)(ap + 32 + lowK), *(const v4u*)(ap + 32 + lowK + 16));
#pragma unroll
    for (int i = 0; i < 4; ++i) {
      const int n = (q + 4 * i) * 16 + col;
      v8f acc = {};
      acc = __builtin_amdgcn_wmma_f32_16x16x32_bf16(false, a0, false, bfr[2*i],     (short)0, acc, false, false);
      acc = __builtin_amdgcn_wmma_f32_16x16x32_bf16(false, a1, false, bfr[2*i + 1], (short)0, acc, false, false);
#pragma unroll
      for (int e = 0; e < 8; ++e) acc[e] = a_act * tanh_f(s_act * acc[e]) + b_act;
      F4x2 s = __builtin_bit_cast(F4x2, acc);
      float* gp = gsh + ((mt * NGP + n) * 16 + r0);
      *(v4f*)gp       = s.lo;
      *(v4f*)(gp + 4) = s.hi;
    }

    // (c) cell update on this wave's own tiles (same-wave LDS order, no barrier)
#pragma unroll
    for (int kk = 0; kk < 8; ++kk) {
      const int p    = lane + 32 * kk;     // 0..255
      const int row  = p & 15;
      const int jloc = p >> 4;             // 0..15
      const int ti   = jloc >> 2;          // which of this wave's 4 tiles
      const int jin  = jloc & 3;
      const int j    = 4 * (q + 4 * ti) + jin;   // hidden index
      if (j < H2) {
        const float* g4 = gm + (4 * j) * 16 + row;
        float gI = g4[0];
        float gF = g4[16];
        float gG = g4[32];
        float gO = g4[48];
        float cn = gF * c[kk] + gI * gG;
        c[kk] = cn;
        float hv = gO * tanh_f(cn);
        bufB[(mt * 16 + row) * KD + 8 + j] = f2bf(hv);   // h_prev for step t+1
        if (t == T - 1) hfin[(size_t)(bbase + mt * 16 + row) * H2 + j] = hv;
      }
    }

    // (a') commit prefetched hs1 row for t+1 into the write buffer
    if (stager && (t + 1 < T)) {
      *(v4u*)(bufB + (mt * 16 + lane) * KD) = nxt;
    }
    __syncthreads();     // single barrier: buf[(t+1)&1] writes -> step t+1 reads
  }
}

// ---------------------------------------------------------------------------
// K3: out = relu(h_last) @ fc_w.T + fc_b
// ---------------------------------------------------------------------------
__global__ void __launch_bounds__(256, 1)
k3_head(const float* __restrict__ hfin,
        const float* __restrict__ fcw,
        const float* __restrict__ fcb,
        float* __restrict__ out) {
  const int i = blockIdx.x * 256 + threadIdx.x;
  if (i >= NB * NC) return;
  const int b = i / NC, n = i % NC;
  float s = fcb[n];
  const float* hp = hfin + (size_t)b * H2;
  const float* wp = fcw + (size_t)n * H2;
#pragma unroll 10
  for (int j = 0; j < H2; ++j) {
    float hv = hp[j];
    hv = hv > 0.0f ? hv : 0.0f;
    s += hv * wp[j];
  }
  out[i] = s;
}

// ---------------------------------------------------------------------------
extern "C" void kernel_launch(void* const* d_in, const int* in_sizes, int n_in,
                              void* d_out, int out_size, void* d_ws, size_t ws_size,
                              hipStream_t stream) {
  (void)in_sizes; (void)n_in; (void)out_size; (void)ws_size;
  const float* x    = (const float*)d_in[0];
  const float* Wih1 = (const float*)d_in[1];
  const float* Whh1 = (const float*)d_in[2];
  const float* bih1 = (const float*)d_in[3];
  const float* bhh1 = (const float*)d_in[4];
  const float* Wih2 = (const float*)d_in[5];
  const float* Whh2 = (const float*)d_in[6];
  const float* bih2 = (const float*)d_in[7];
  const float* bhh2 = (const float*)d_in[8];
  const float* fcw  = (const float*)d_in[9];
  const float* fcb  = (const float*)d_in[10];
  float* out = (float*)d_out;

  char* ws = (char*)d_ws;
  const size_t xp1_bytes  = (size_t)T * NB * XPR * 2;   // 48 MB
  const size_t hs1e_bytes = (size_t)T * NB * 8 * 2;     // 16 MB
  unsigned short* xp1  = (unsigned short*)ws;
  unsigned short* hs1e = (unsigned short*)(ws + xp1_bytes);
  float*          hfin = (float*)(ws + xp1_bytes + hs1e_bytes);

  k0_inproj<<<(T * NB) / 256, 256, 0, stream>>>(x, Wih1, bih1, bhh1, xp1);
  k1_scan1 <<<4, 128, 0, stream>>>(Whh1, xp1, hs1e);
  const size_t smem = (size_t)NGP * KD * 2 + 2 * ABUF * 2 + (size_t)4 * NGP * 16 * 4; // 114688 B
  k2_scan2 <<<8, 512, smem, stream>>>(Wih2, Whh2, bih2, bhh2, hs1e, hfin);
  k3_head  <<<(NB * NC + 255) / 256, 256, 0, stream>>>(hfin, fcw, fcb, out);
}
